// SelfAttentionModule_65481071400267
// MI455X (gfx1250) — compile-verified, matrix operand order checked by reading, not run on hardware
//
#include <hip/hip_runtime.h>
#include <hip/hip_bf16.h>

// ---------------- Types for WMMA ----------------
typedef __bf16 bf16_t;
typedef __attribute__((ext_vector_type(16))) __bf16        v16bf;
typedef __attribute__((ext_vector_type(8)))  float         v8f;
typedef __attribute__((ext_vector_type(4)))  unsigned int  v4u;

union TileU { v16bf v; v4u u[2]; };

#define DIM   1024
#define NH    16
#define HD    64
#define TT    2048
#define BB    4
#define MM    (BB*TT)   // 8192 rows

#if __has_builtin(__builtin_amdgcn_tensor_load_to_lds)
#define HAVE_TDM 1
#else
#define HAVE_TDM 0
#endif

// round-to-nearest-even f32 -> bf16 (as raw u16)
__device__ __forceinline__ unsigned short f2bf(float f) {
    unsigned int u = __float_as_uint(f);
    unsigned int r = u + 0x7FFFu + ((u >> 16) & 1u);
    return (unsigned short)(r >> 16);
}

// Load one 16(M/N) x 32(K) row-major bf16 tile into WMMA A/B operand layout.
// Lanes 0-15: row = lane,    K = {0..7} in V0-3 and {16..23} in V4-7
// Lanes 16-31: row = lane-16, K = {8..15, 24..31}
__device__ __forceinline__ v16bf load_tile16x32(const unsigned short* p, int ld) {
    int lane = threadIdx.x & 31;
    int row  = lane & 15;
    int kb   = (lane >> 4) << 3;        // 0 or 8 elements
    const unsigned short* r0 = p + (size_t)row * ld + kb;
    TileU t;
    t.u[0] = *(const v4u*)(r0);         // K = kb .. kb+7
    t.u[1] = *(const v4u*)(r0 + 16);    // K = kb+16 .. kb+23
    return t.v;
}

__device__ __forceinline__ v8f wmma_bf16(v16bf a, v16bf b, v8f c) {
    return __builtin_amdgcn_wmma_f32_16x16x32_bf16(
        /*neg_a=*/false, a, /*neg_b=*/false, b,
        /*c_mod=*/(short)0, c, /*reuse_a=*/false, /*reuse_b=*/false);
}

// low 32 bits of a generic pointer to LDS = LDS byte address
__device__ __forceinline__ unsigned lds_offset(const void* p) {
    return (unsigned)(unsigned long long)p;
}

#if HAVE_TDM
typedef unsigned int u32x4 __attribute__((ext_vector_type(4)));
typedef int          i32x8 __attribute__((ext_vector_type(8)));
typedef int          i32x4 __attribute__((ext_vector_type(4)));

// Issue one TDM 2D tile load: rows x cols bf16 tile (row stride ldElems) -> LDS
// (packed row-major, rows*cols*2 bytes). D# built per CDNA5 ISA 8.3/8.4.
// This toolchain's builtin takes 6 args: (g0 u32x4, g1 i32x8, g2 i32x4, g3 i32x4, extra i32x8, cpol)
__device__ __forceinline__ void tdm_issue_2d(const unsigned short* g, unsigned ldsOff,
                                             unsigned rows, unsigned cols,
                                             unsigned ldElems) {
    unsigned long long ga = (unsigned long long)g;
    u32x4 g0;
    g0[0] = 1u;                                                  // count=1, user D#
    g0[1] = ldsOff;                                              // lds_addr (bytes)
    g0[2] = (unsigned)ga;                                        // global_addr[31:0]
    g0[3] = (unsigned)((ga >> 32) & 0x01FFFFFFu) | 0x80000000u;  // addr[56:32] | type=2
    const unsigned td0 = 0x40000000u, td1 = 0x40000000u;         // huge tensor dims (tile in-bounds)
    i32x8 g1;
    g1[0] = (int)(1u << 16);                                     // data_size=1 -> 2 bytes/elem
    g1[1] = (int)((td0 & 0xFFFFu) << 16);                        // tensor_dim0[15:0]
    g1[2] = (int)(((td0 >> 16) & 0xFFFFu) | ((td1 & 0xFFFFu) << 16));
    g1[3] = (int)(((td1 >> 16) & 0xFFFFu) | ((cols & 0xFFFFu) << 16)); // tile_dim0
    g1[4] = (int)(rows & 0xFFFFu);                               // tile_dim1 (tile_dim2=0)
    g1[5] = (int)ldElems;                                        // tensor_dim0_stride[31:0]
    g1[6] = 0;
    g1[7] = 0;
    i32x4 z4 = {0, 0, 0, 0};
    i32x8 z8 = {0, 0, 0, 0, 0, 0, 0, 0};
    __builtin_amdgcn_tensor_load_to_lds(g0, g1, z4, z4, z8, 0);
}
#else
// Fallback: cooperative 64x32 bf16 tile copy by 128 threads
__device__ __forceinline__ void coop_copy64x32(unsigned short* dst,
                                               const unsigned short* src, int ld) {
    int tid = threadIdx.x;                 // 0..127
    #pragma unroll
    for (int idx = tid; idx < 256; idx += 128) {
        int row = idx >> 2;
        int c   = (idx & 3) * 8;
        *(v4u*)(dst + row * 32 + c) = *(const v4u*)(src + (size_t)row * ld + c);
    }
}
#endif

// ---------------- Kernel 1: f32 -> bf16 elementwise ----------------
__global__ void cvt_bf16_kernel(const float* __restrict__ in,
                                unsigned short* __restrict__ out, int n) {
    int i = blockIdx.x * blockDim.x + threadIdx.x;
    if (i < n) out[i] = f2bf(in[i]);
}

// ---------------- Kernel 2: transpose DxD f32 -> bf16 (Wt[n][k] = W[k][n]) ----
__global__ void transpose_bf16_kernel(const float* __restrict__ w,
                                      unsigned short* __restrict__ wt) {
    int n = blockIdx.x * 16 + threadIdx.x;
    int k = blockIdx.y * 16 + threadIdx.y;
    wt[(size_t)n * DIM + k] = f2bf(w[(size_t)k * DIM + n]);
}

// ---------------- Kernel 3: WMMA GEMM with TDM double-buffered LDS staging ---
// A: bf16 row-major MxK;  Bt: bf16 row-major NxK (B transposed)
// 4 waves / block; block tile 64(M) x 64(N); K-chunk 32, double buffered.
// mode 0: out f32 row-major MxN
// mode 1: out bf16 [b][h][t][hd]    (Q / K layout)
// mode 2: out bf16 [b][h][hd][t]    (V transposed layout)
__global__ __launch_bounds__(128) void gemm_wmma_kernel(
        const unsigned short* __restrict__ A,
        const unsigned short* __restrict__ Bt,
        const float* __restrict__ bias,
        void* __restrict__ out,
        int M, int N, int Ktot, int mode) {
    __shared__ __align__(16) unsigned short Abuf[2][64 * 32];
    __shared__ __align__(16) unsigned short Bbuf[2][64 * 32];

    int m0   = blockIdx.x * 64;
    int n0   = blockIdx.y * 64;
    int wave = threadIdx.x >> 5;
    int nch  = Ktot >> 5;
    const unsigned short* Ab = A  + (size_t)m0 * Ktot;
    const unsigned short* Bb = Bt + (size_t)n0 * Ktot;

    v8f acc0 = {}, acc1 = {}, acc2 = {}, acc3 = {};

    // ---- stage chunk 0 ----
#if HAVE_TDM
    if (threadIdx.x == 0) {
        tdm_issue_2d(Ab, lds_offset(Abuf[0]), 64, 32, Ktot);
        tdm_issue_2d(Bb, lds_offset(Bbuf[0]), 64, 32, Ktot);
    }
#else
    coop_copy64x32(Abuf[0], Ab, Ktot);
    coop_copy64x32(Bbuf[0], Bb, Ktot);
#endif

    for (int i = 0; i < nch; ++i) {
        int cur = i & 1;
        // ---- stage chunk i+1 into the other buffer, then ensure chunk i done ----
#if HAVE_TDM
        if (threadIdx.x == 0) {
            if (i + 1 < nch) {
                tdm_issue_2d(Ab + (size_t)(i + 1) * 32, lds_offset(Abuf[cur ^ 1]), 64, 32, Ktot);
                tdm_issue_2d(Bb + (size_t)(i + 1) * 32, lds_offset(Bbuf[cur ^ 1]), 64, 32, Ktot);
                __builtin_amdgcn_s_wait_tensorcnt(2);   // chunk i's 2 TDM ops retired
            } else {
                __builtin_amdgcn_s_wait_tensorcnt(0);
            }
        }
#else
        if (i + 1 < nch) {
            coop_copy64x32(Abuf[cur ^ 1], Ab + (size_t)(i + 1) * 32, Ktot);
            coop_copy64x32(Bbuf[cur ^ 1], Bb + (size_t)(i + 1) * 32, Ktot);
        }
#endif
        __syncthreads();                       // chunk i visible to all waves

        // ---- compute on chunk i from LDS (ds_load_b128) ----
        v16bf a  = load_tile16x32(&Abuf[cur][wave * 16 * 32], 32);
        v16bf b0 = load_tile16x32(&Bbuf[cur][0 * 16 * 32], 32);
        v16bf b1 = load_tile16x32(&Bbuf[cur][1 * 16 * 32], 32);
        v16bf b2 = load_tile16x32(&Bbuf[cur][2 * 16 * 32], 32);
        v16bf b3 = load_tile16x32(&Bbuf[cur][3 * 16 * 32], 32);
        acc0 = wmma_bf16(a, b0, acc0);
        acc1 = wmma_bf16(a, b1, acc1);
        acc2 = wmma_bf16(a, b2, acc2);
        acc3 = wmma_bf16(a, b3, acc3);
        __syncthreads();                       // all waves done reading buf[cur]
    }

    // ---- epilogue: lane holds col n = n_tile + (lane&15); VGPR r -> row +r+8*(lane>>4)
    int lane  = threadIdx.x & 31;
    int nl    = lane & 15;
    int half  = lane >> 4;
    int mbase = m0 + wave * 16;
    v8f accs[4] = {acc0, acc1, acc2, acc3};
    #pragma unroll
    for (int j = 0; j < 4; ++j) {
        int n = n0 + j * 16 + nl;
        float bv = bias[n];
        #pragma unroll
        for (int r = 0; r < 8; ++r) {
            int m = mbase + r + 8 * half;
            float val = accs[j][r] + bv;
            if (mode == 0) {
                ((float*)out)[(size_t)m * N + n] = val;
            } else {
                int b  = m >> 11;        // m / 2048
                int t  = m & (TT - 1);
                int h  = n >> 6;         // n / 64
                int hd = n & (HD - 1);
                size_t idx;
                if (mode == 1) idx = (((size_t)(b * NH + h)) * TT + t) * HD + hd;
                else           idx = (((size_t)(b * NH + h)) * HD + hd) * TT + t;
                ((unsigned short*)out)[idx] = f2bf(val);
            }
        }
    }
}

// ---------------- Kernel 4: flash attention (1 wave per 16-query tile) -------
// Q,K: bf16 [b][h][t][hd]   Vt: bf16 [b][h][hd][t]   Y: bf16 [b*T][D] row-major
__global__ void attn_wmma_kernel(const unsigned short* __restrict__ Q,
                                 const unsigned short* __restrict__ K,
                                 const unsigned short* __restrict__ Vt,
                                 unsigned short* __restrict__ Y) {
    __shared__ __align__(16) unsigned short Plds[16 * 32];   // P tile (16q x 32k) bf16
    int bh = blockIdx.x;                 // b*NH + h
    int qt = blockIdx.y;                 // query tile index
    int lane = threadIdx.x & 31;
    int half = lane >> 4;

    const unsigned short* Qb = Q  + (((size_t)bh * TT) + (size_t)qt * 16) * HD;
    const unsigned short* Kb = K  + (size_t)bh * TT * HD;
    const unsigned short* Vb = Vt + (size_t)bh * HD * TT;

    v16bf qa0 = load_tile16x32(Qb + 0,  HD);     // hd 0..31
    v16bf qa1 = load_tile16x32(Qb + 32, HD);     // hd 32..63

    v8f o0 = {}, o1 = {}, o2 = {}, o3 = {};
    float mrun[8], lrun[8];
    #pragma unroll
    for (int r = 0; r < 8; ++r) { mrun[r] = -1e30f; lrun[r] = 0.f; }

    for (int kc = 0; kc < TT; kc += 32) {
        if (kc + 32 < TT) __builtin_prefetch(Kb + (size_t)(kc + 32) * HD, 0, 0);
        // ---- scores: S = (Q @ K^T) * 1/sqrt(Hd) ----
        v16bf b00 = load_tile16x32(Kb + (size_t)(kc +  0) * HD +  0, HD);
        v16bf b01 = load_tile16x32(Kb + (size_t)(kc +  0) * HD + 32, HD);
        v16bf b10 = load_tile16x32(Kb + (size_t)(kc + 16) * HD +  0, HD);
        v16bf b11 = load_tile16x32(Kb + (size_t)(kc + 16) * HD + 32, HD);
        v8f s0 = {}, s1 = {};
        s0 = wmma_bf16(qa0, b00, s0);  s0 = wmma_bf16(qa1, b01, s0);
        s1 = wmma_bf16(qa0, b10, s1);  s1 = wmma_bf16(qa1, b11, s1);
        #pragma unroll
        for (int r = 0; r < 8; ++r) { s0[r] *= 0.125f; s1[r] *= 0.125f; }

        // ---- online softmax (row m = r + 8*half lives in VGPR r of this half) ----
        float nm[8], scl[8], p0[8], p1[8];
        #pragma unroll
        for (int r = 0; r < 8; ++r) {
            float cm = fmaxf(s0[r], s1[r]);
            cm = fmaxf(cm, __shfl_xor(cm, 1, 32));
            cm = fmaxf(cm, __shfl_xor(cm, 2, 32));
            cm = fmaxf(cm, __shfl_xor(cm, 4, 32));
            cm = fmaxf(cm, __shfl_xor(cm, 8, 32));   // masks<16 stay inside 16-lane half
            nm[r] = fmaxf(mrun[r], cm);
        }
        #pragma unroll
        for (int r = 0; r < 8; ++r) {
            p0[r] = __expf(s0[r] - nm[r]);
            p1[r] = __expf(s1[r] - nm[r]);
            float cs = p0[r] + p1[r];
            cs += __shfl_xor(cs, 1, 32);
            cs += __shfl_xor(cs, 2, 32);
            cs += __shfl_xor(cs, 4, 32);
            cs += __shfl_xor(cs, 8, 32);
            scl[r]  = __expf(mrun[r] - nm[r]);
            lrun[r] = lrun[r] * scl[r] + cs;
            mrun[r] = nm[r];
        }
        #pragma unroll
        for (int r = 0; r < 8; ++r) {
            o0[r] *= scl[r]; o1[r] *= scl[r]; o2[r] *= scl[r]; o3[r] *= scl[r];
        }

        // ---- C-layout -> A-layout via LDS ----
        #pragma unroll
        for (int r = 0; r < 8; ++r) {
            int m = r + 8 * half;
            Plds[m * 32 + (lane & 15)]      = f2bf(p0[r]);
            Plds[m * 32 + 16 + (lane & 15)] = f2bf(p1[r]);
        }
        __syncthreads();
        v16bf pa = load_tile16x32(Plds, 32);

        // ---- O += P @ V (Vt rows = hd, cols = t) ----
        v16bf v0 = load_tile16x32(Vb + (size_t)( 0) * TT + kc, TT);
        v16bf v1 = load_tile16x32(Vb + (size_t)(16) * TT + kc, TT);
        v16bf v2 = load_tile16x32(Vb + (size_t)(32) * TT + kc, TT);
        v16bf v3 = load_tile16x32(Vb + (size_t)(48) * TT + kc, TT);
        o0 = wmma_bf16(pa, v0, o0);
        o1 = wmma_bf16(pa, v1, o1);
        o2 = wmma_bf16(pa, v2, o2);
        o3 = wmma_bf16(pa, v3, o3);
        __syncthreads();
    }

    // ---- finalize: divide by row sums, write Y bf16 row-major (B*T, D) ----
    int b = bh >> 4;
    int h = bh & (NH - 1);
    float inv[8];
    #pragma unroll
    for (int r = 0; r < 8; ++r) inv[r] = 1.0f / lrun[r];
    v8f os[4] = {o0, o1, o2, o3};
    #pragma unroll
    for (int j = 0; j < 4; ++j) {
        #pragma unroll
        for (int r = 0; r < 8; ++r) {
            int m   = qt * 16 + r + 8 * half;
            int col = h * HD + j * 16 + (lane & 15);
            Y[((size_t)b * TT + m) * DIM + col] = f2bf(os[j][r] * inv[r]);
        }
    }
}

// ---------------- Kernel 5: in-place LayerNorm over rows of d_out ----------
__global__ void layernorm_kernel(float* __restrict__ z,
                                 const float* __restrict__ gamma,
                                 const float* __restrict__ beta) {
    __shared__ float row[DIM];
    __shared__ float red[256];
    int r   = blockIdx.x;
    int tid = threadIdx.x;
    float* zr = z + (size_t)r * DIM;
    float s = 0.f;
    for (int i = tid; i < DIM; i += 256) { float v = zr[i]; row[i] = v; s += v; }
    red[tid] = s; __syncthreads();
    for (int st = 128; st > 0; st >>= 1) {
        if (tid < st) red[tid] += red[tid + st];
        __syncthreads();
    }
    float mu = red[0] / (float)DIM;
    __syncthreads();
    float q = 0.f;
    for (int i = tid; i < DIM; i += 256) { float d = row[i] - mu; q += d * d; }
    red[tid] = q; __syncthreads();
    for (int st = 128; st > 0; st >>= 1) {
        if (tid < st) red[tid] += red[tid + st];
        __syncthreads();
    }
    float invs = rsqrtf(red[0] / (float)DIM + 1e-5f);
    for (int i = tid; i < DIM; i += 256)
        zr[i] = (row[i] - mu) * invs * gamma[i] + beta[i];
}

// ---------------- Host launch ----------------
extern "C" void kernel_launch(void* const* d_in, const int* in_sizes, int n_in,
                              void* d_out, int out_size, void* d_ws, size_t ws_size,
                              hipStream_t stream) {
    const float* x   = (const float*)d_in[0];
    const float* wq  = (const float*)d_in[1];
    const float* bq  = (const float*)d_in[2];
    const float* wk  = (const float*)d_in[3];
    const float* bk  = (const float*)d_in[4];
    const float* wv  = (const float*)d_in[5];
    const float* bv  = (const float*)d_in[6];
    const float* wo  = (const float*)d_in[7];
    const float* bo  = (const float*)d_in[8];
    const float* lng = (const float*)d_in[9];
    const float* lnb = (const float*)d_in[10];
    float* out = (float*)d_out;

    // workspace layout (bf16 as u16)
    unsigned short* ws  = (unsigned short*)d_ws;
    unsigned short* Xb  = ws;                               // 8192x1024
    unsigned short* WqT = Xb  + (size_t)MM * DIM;           // 1024x1024 each
    unsigned short* WkT = WqT + (size_t)DIM * DIM;
    unsigned short* WvT = WkT + (size_t)DIM * DIM;
    unsigned short* WoT = WvT + (size_t)DIM * DIM;
    unsigned short* Qb  = WoT + (size_t)DIM * DIM;          // [b][h][t][hd]
    unsigned short* Kb  = Qb  + (size_t)MM * DIM;
    unsigned short* Vt  = Kb  + (size_t)MM * DIM;           // [b][h][hd][t]
    unsigned short* Yb  = Vt  + (size_t)MM * DIM;           // [b*t][D]

    // 1) convert x -> bf16
    {
        int n = MM * DIM;
        cvt_bf16_kernel<<<(n + 255) / 256, 256, 0, stream>>>(x, Xb, n);
    }
    // 2) transpose weights -> bf16 Wt[n][k]
    {
        dim3 g(DIM / 16, DIM / 16), blk(16, 16);
        transpose_bf16_kernel<<<g, blk, 0, stream>>>(wq, WqT);
        transpose_bf16_kernel<<<g, blk, 0, stream>>>(wk, WkT);
        transpose_bf16_kernel<<<g, blk, 0, stream>>>(wv, WvT);
        transpose_bf16_kernel<<<g, blk, 0, stream>>>(wo, WoT);
    }
    // 3) QKV projections (WMMA + TDM staging)
    {
        dim3 g(MM / 64, DIM / 64);
        gemm_wmma_kernel<<<g, 128, 0, stream>>>(Xb, WqT, bq, Qb, MM, DIM, DIM, 1);
        gemm_wmma_kernel<<<g, 128, 0, stream>>>(Xb, WkT, bk, Kb, MM, DIM, DIM, 1);
        gemm_wmma_kernel<<<g, 128, 0, stream>>>(Xb, WvT, bv, Vt, MM, DIM, DIM, 2);
    }
    // 4) flash attention (WMMA)
    {
        dim3 g(BB * NH, TT / 16);
        attn_wmma_kernel<<<g, 32, 0, stream>>>(Qb, Kb, Vt, Yb);
    }
    // 5) output projection -> f32 d_out (WMMA + TDM staging)
    {
        dim3 g(MM / 64, DIM / 64);
        gemm_wmma_kernel<<<g, 128, 0, stream>>>(Yb, WoT, bo, out, MM, DIM, DIM, 0);
    }
    // 6) LayerNorm in place on d_out
    layernorm_kernel<<<MM, 256, 0, stream>>>(out, lng, lnb);
}
